// LinearToeplitz_2173253452210
// MI455X (gfx1250) — compile-verified, matrix operand order checked by reading, not run on hardware
//
#include <hip/hip_runtime.h>

// LinearToeplitz: out[b,o] = sum_i x[b,i] * p[(O-1)-o+i] + bias[o]
// GEMM M=8192 N=4096 K=4096 (~275 GFLOP, compute bound).
//
// Fast path (workspace >= 64MB+16KB):
//   1) one-time convert X and params to bf16 in d_ws (bandwidth-trivial)
//   2) WMMA GEMM with BOTH bf16 tiles staged by the Tensor Data Mover:
//        A: 128x32 bf16 tile of X          (row stride 4096)
//        B: Toeplitz params window, row-flipped so row stride == 1
//      -> hot loop is ds_load_b128 + v_wmma only, no conversions.
// Fallback path: same pipeline but TDM stages f32 and fragments convert
// with v_cvt_pk_bf16_f32 (no workspace needed).

#define MDIM 8192
#define NDIM 4096
#define KDIM 4096
#define PLEN (NDIM + KDIM - 1) // 8191

#define BM 128
#define BN 128
#define BK 32
#define TILE_ELEMS (128 * 32)

typedef __attribute__((ext_vector_type(16))) __bf16 v16bf;
typedef __attribute__((ext_vector_type(8)))  float  v8f;
typedef __attribute__((ext_vector_type(4)))  unsigned int u32x4;
typedef __attribute__((ext_vector_type(8)))  int          i32x8;
typedef __attribute__((ext_vector_type(4)))  int          i32x4;

union FragBF { __bf16 h[16]; uint4 q[2]; v16bf v; };
union H8     { __bf16 h[8];  uint4 q; };

// ---- TDM: one 2D tile (tile_dim0=BK elems, tile_dim1=BM rows) ----------
// D# packing per CDNA5 ISA §8.3/§8.4 (groups 2/3 zero => 2D tensor).
// ds_code: 1 -> 2-byte elements (bf16), 2 -> 4-byte elements (f32).
__device__ __forceinline__ void tdm_load_tile(unsigned lds_byte_off,
                                              const void* gsrc,
                                              unsigned ds_code,
                                              unsigned tensor_d0,
                                              unsigned tensor_d1,
                                              unsigned long long stride0)
{
  unsigned long long ga = (unsigned long long)(uintptr_t)gsrc;
  u32x4 g0;
  g0[0] = 1u;                                   // count=1 (valid), user mode
  g0[1] = lds_byte_off;                         // lds_addr [63:32]
  g0[2] = (unsigned)ga;                         // global_addr [95:64]
  g0[3] = (unsigned)((ga >> 32) & 0x01FFFFFFu)  // global_addr [120:96]
        | (2u << 30);                           // type=2 ("image") [127:126]
  i32x8 g1;
  g1[0] = (int)(ds_code << 16);                 // data_size
  g1[1] = (int)((tensor_d0 & 0xFFFFu) << 16);   // tensor_dim0[15:0] @ bits 63:48
  g1[2] = (int)((tensor_d0 >> 16) | ((tensor_d1 & 0xFFFFu) << 16));
  g1[3] = (int)((tensor_d1 >> 16) | ((unsigned)BK << 16));   // tile_dim0=32
  g1[4] = (int)(unsigned)BM;                    // tile_dim1=128, tile_dim2=0
  g1[5] = (int)(unsigned)(stride0 & 0xFFFFFFFFull);          // dim0 stride lo
  g1[6] = (int)(unsigned)((stride0 >> 32) & 0xFFFFull);      // stride hi
  g1[7] = 0;
  i32x4 z4 = {0, 0, 0, 0};
  i32x8 z8 = {0, 0, 0, 0, 0, 0, 0, 0};
  __builtin_amdgcn_tensor_load_to_lds(g0, g1, z4, z4, z8, 0);
}

// bf16 fragment from a bf16 LDS tile row (K runs {b..b+7},{b+16..b+23})
__device__ __forceinline__ v16bf frag_bf16(const __bf16* t, int off0)
{
  const uint4* p = (const uint4*)t; // 8 bf16 per uint4
  FragBF f;
  f.q[0] = p[off0 >> 3];
  f.q[1] = p[(off0 + 16) >> 3];
  return f.v;
}

// bf16 fragment from an f32 LDS tile row (fallback path)
__device__ __forceinline__ v16bf frag_f32(const float* t, int off0)
{
  const float4* p = (const float4*)t;
  float4 a = p[off0 >> 2];
  float4 b = p[(off0 + 4) >> 2];
  float4 c = p[(off0 + 16) >> 2];
  float4 d = p[(off0 + 20) >> 2];
  FragBF f;
  f.h[0]  = (__bf16)a.x; f.h[1]  = (__bf16)a.y; f.h[2]  = (__bf16)a.z; f.h[3]  = (__bf16)a.w;
  f.h[4]  = (__bf16)b.x; f.h[5]  = (__bf16)b.y; f.h[6]  = (__bf16)b.z; f.h[7]  = (__bf16)b.w;
  f.h[8]  = (__bf16)c.x; f.h[9]  = (__bf16)c.y; f.h[10] = (__bf16)c.z; f.h[11] = (__bf16)c.w;
  f.h[12] = (__bf16)d.x; f.h[13] = (__bf16)d.y; f.h[14] = (__bf16)d.z; f.h[15] = (__bf16)d.w;
  return f.v;
}

// ---- precompute: X f32 -> bf16 (one 8-elem chunk per thread, no loop) ------------
__global__ void cvt_x_bf16(const float* __restrict__ x, __bf16* __restrict__ xb)
{
  const size_t i = ((size_t)blockIdx.x * blockDim.x + threadIdx.x) * 8;
  const float4* p = (const float4*)(x + i);
  float4 a = p[0], b = p[1];
  H8 o;
  o.h[0] = (__bf16)a.x; o.h[1] = (__bf16)a.y; o.h[2] = (__bf16)a.z; o.h[3] = (__bf16)a.w;
  o.h[4] = (__bf16)b.x; o.h[5] = (__bf16)b.y; o.h[6] = (__bf16)b.z; o.h[7] = (__bf16)b.w;
  *(uint4*)(xb + i) = o.q;
}

__global__ void cvt_p_bf16(const float* __restrict__ p, __bf16* __restrict__ pb)
{
  for (int i = threadIdx.x; i < PLEN + 1; i += blockDim.x)
    pb[i] = (i < PLEN) ? (__bf16)p[i] : (__bf16)0.0f;
}

// ======================= fast path: bf16 TDM GEMM ========================
__launch_bounds__(256)
__global__ void toeplitz_gemm_wmma_tdm_bf16(const __bf16* __restrict__ xb,
                                            const __bf16* __restrict__ pb,
                                            const float* __restrict__ bias,
                                            float* __restrict__ out)
{
  __shared__ __align__(16) __bf16 LA[2][TILE_ELEMS]; // 2 x 8 KB
  __shared__ __align__(16) __bf16 LB[2][TILE_ELEMS]; // 2 x 8 KB

  const int tid = threadIdx.x;
  const int n0  = blockIdx.x * BN;
  const int m0  = blockIdx.y * BM;

  const int wid  = tid >> 5;
  const int lane = tid & 31;
  const int lr   = lane & 15;
  const int hi   = lane >> 4;
  const int kb8  = hi * 8;            // per-lane K base (ISA §7.12.2)
  const int rm   = (wid & 3) * 32;
  const int cn   = (wid >> 2) * 64;

  const unsigned ldsA = (unsigned)(uintptr_t)(&LA[0][0]);
  const unsigned ldsB = (unsigned)(uintptr_t)(&LB[0][0]);
  const int pbase0 = (NDIM - BN) - n0; // row-flipped Toeplitz window base

  v8f acc[2][4];
#pragma unroll
  for (int mt = 0; mt < 2; ++mt)
#pragma unroll
    for (int nt = 0; nt < 4; ++nt)
      acc[mt][nt] = 0.0f;

  if (tid < 32) {
    tdm_load_tile(ldsA, xb + (size_t)m0 * KDIM, 1, KDIM, MDIM, KDIM);
    tdm_load_tile(ldsB, pb + pbase0, 1, PLEN, PLEN, 1);
  }

  for (int k0 = 0; k0 < KDIM; k0 += BK) {
    const int buf = (k0 >> 5) & 1;

    if (tid < 32) __builtin_amdgcn_s_wait_tensorcnt(0);
    __syncthreads();

    if ((tid < 32) && (k0 + BK < KDIM)) {
      const int kn = k0 + BK;
      const unsigned off = (unsigned)((buf ^ 1) * TILE_ELEMS * 2);
      tdm_load_tile(ldsA + off, xb + (size_t)m0 * KDIM + kn, 1, KDIM, MDIM, KDIM);
      tdm_load_tile(ldsB + off, pb + (pbase0 + kn), 1, PLEN, PLEN, 1);
    }

    const __bf16* At = LA[buf];
    const __bf16* Bt = LB[buf];

    v16bf fa[2];
#pragma unroll
    for (int mt = 0; mt < 2; ++mt)
      fa[mt] = frag_bf16(At, (rm + mt * 16 + lr) * BK + kb8);

    v16bf fb[4];
#pragma unroll
    for (int nt = 0; nt < 4; ++nt) {
      const int rowL = (BN - 1) - (cn + nt * 16 + lr);
      fb[nt] = frag_bf16(Bt, rowL * BK + kb8);
    }

#pragma unroll
    for (int nt = 0; nt < 4; ++nt)
#pragma unroll
      for (int mt = 0; mt < 2; ++mt)
        acc[mt][nt] = __builtin_amdgcn_wmma_f32_16x16x32_bf16(
            false, fa[mt], false, fb[nt], (short)0, acc[mt][nt], false, false);
  }

  float bv[4];
#pragma unroll
  for (int nt = 0; nt < 4; ++nt) bv[nt] = bias[n0 + cn + nt * 16 + lr];

#pragma unroll
  for (int mt = 0; mt < 2; ++mt) {
#pragma unroll
    for (int r = 0; r < 8; ++r) {
      const size_t row = (size_t)(m0 + rm + mt * 16 + hi * 8 + r);
      float* orow = out + row * NDIM + (n0 + cn + lr);
#pragma unroll
      for (int nt = 0; nt < 4; ++nt)
        orow[nt * 16] = acc[mt][nt][r] + bv[nt];
    }
  }
}

// ================= fallback: f32 TDM GEMM (no workspace) =================
__launch_bounds__(256)
__global__ void toeplitz_gemm_wmma_tdm_f32(const float* __restrict__ x,
                                           const float* __restrict__ params,
                                           const float* __restrict__ bias,
                                           float* __restrict__ out)
{
  __shared__ __align__(16) float LA[2][TILE_ELEMS]; // 2 x 16 KB
  __shared__ __align__(16) float LB[2][TILE_ELEMS]; // 2 x 16 KB

  const int tid = threadIdx.x;
  const int n0  = blockIdx.x * BN;
  const int m0  = blockIdx.y * BM;

  const int wid  = tid >> 5;
  const int lane = tid & 31;
  const int lr   = lane & 15;
  const int hi   = lane >> 4;
  const int kb8  = hi * 8;
  const int rm   = (wid & 3) * 32;
  const int cn   = (wid >> 2) * 64;

  const unsigned ldsA = (unsigned)(uintptr_t)(&LA[0][0]);
  const unsigned ldsB = (unsigned)(uintptr_t)(&LB[0][0]);
  const int pbase0 = (NDIM - BN) - n0;

  v8f acc[2][4];
#pragma unroll
  for (int mt = 0; mt < 2; ++mt)
#pragma unroll
    for (int nt = 0; nt < 4; ++nt)
      acc[mt][nt] = 0.0f;

  if (tid < 32) {
    tdm_load_tile(ldsA, x + (size_t)m0 * KDIM, 2, KDIM, MDIM, KDIM);
    tdm_load_tile(ldsB, params + pbase0, 2, PLEN, PLEN, 1);
  }

  for (int k0 = 0; k0 < KDIM; k0 += BK) {
    const int buf = (k0 >> 5) & 1;

    if (tid < 32) __builtin_amdgcn_s_wait_tensorcnt(0);
    __syncthreads();

    if ((tid < 32) && (k0 + BK < KDIM)) {
      const int kn = k0 + BK;
      const unsigned off = (unsigned)((buf ^ 1) * TILE_ELEMS * 4);
      tdm_load_tile(ldsA + off, x + (size_t)m0 * KDIM + kn, 2, KDIM, MDIM, KDIM);
      tdm_load_tile(ldsB + off, params + (pbase0 + kn), 2, PLEN, PLEN, 1);
    }

    const float* At = LA[buf];
    const float* Bt = LB[buf];

    v16bf fa[2];
#pragma unroll
    for (int mt = 0; mt < 2; ++mt)
      fa[mt] = frag_f32(At, (rm + mt * 16 + lr) * BK + kb8);

    v16bf fb[4];
#pragma unroll
    for (int nt = 0; nt < 4; ++nt) {
      const int rowL = (BN - 1) - (cn + nt * 16 + lr);
      fb[nt] = frag_f32(Bt, rowL * BK + kb8);
    }

#pragma unroll
    for (int nt = 0; nt < 4; ++nt)
#pragma unroll
      for (int mt = 0; mt < 2; ++mt)
        acc[mt][nt] = __builtin_amdgcn_wmma_f32_16x16x32_bf16(
            false, fa[mt], false, fb[nt], (short)0, acc[mt][nt], false, false);
  }

  float bv[4];
#pragma unroll
  for (int nt = 0; nt < 4; ++nt) bv[nt] = bias[n0 + cn + nt * 16 + lr];

#pragma unroll
  for (int mt = 0; mt < 2; ++mt) {
#pragma unroll
    for (int r = 0; r < 8; ++r) {
      const size_t row = (size_t)(m0 + rm + mt * 16 + hi * 8 + r);
      float* orow = out + row * NDIM + (n0 + cn + lr);
#pragma unroll
      for (int nt = 0; nt < 4; ++nt)
        orow[nt * 16] = acc[mt][nt][r] + bv[nt];
    }
  }
}

extern "C" void kernel_launch(void* const* d_in, const int* in_sizes, int n_in,
                              void* d_out, int out_size, void* d_ws, size_t ws_size,
                              hipStream_t stream) {
  (void)in_sizes; (void)n_in; (void)out_size;
  const float* x      = (const float*)d_in[0]; // [8192, 4096] f32
  const float* params = (const float*)d_in[1]; // [8191] f32
  const float* bias   = (const float*)d_in[2]; // [4096] f32
  float* out          = (float*)d_out;         // [8192, 4096] f32

  dim3 grid(NDIM / BN, MDIM / BM); // 32 x 64 = 2048 workgroups
  dim3 block(256);                 // 8 waves (wave32)

  const size_t xbytes = (size_t)MDIM * KDIM * sizeof(__bf16); // 64 MB
  const size_t pbytes = (size_t)(PLEN + 1) * sizeof(__bf16);  // 16 KB
  if (d_ws && ws_size >= xbytes + pbytes) {
    __bf16* xb = (__bf16*)d_ws;
    __bf16* pb = (__bf16*)((char*)d_ws + xbytes);
    cvt_x_bf16<<<(MDIM * KDIM) / (256 * 8), 256, 0, stream>>>(x, xb);
    cvt_p_bf16<<<1, 256, 0, stream>>>(params, pb);
    toeplitz_gemm_wmma_tdm_bf16<<<grid, block, 0, stream>>>(xb, pb, bias, out);
  } else {
    toeplitz_gemm_wmma_tdm_f32<<<grid, block, 0, stream>>>(x, params, bias, out);
  }
}
